// Node_encoder_33895881900470
// MI455X (gfx1250) — matrix operand order repaired and checked by reading, not compiled, not run on hardware
//
#include <hip/hip_runtime.h>

// ---------------------------------------------------------------------------
// Node encoder (bi-LSTM x2 + MLP VAE head) for MI455X / gfx1250, wave32 WMMA.
// lstm2 recurrence: v_wmma_f32_16x16x32_f16, operands pre-swizzled into WMMA
// register layout; time-invariant weight B-tiles hoisted into VGPRs; all
// nonlinearities on the straight-line v_exp_f32 path (no libm branches).
// ---------------------------------------------------------------------------

#define NT 160           // sequence length (= N)
#define NB 8             // batch
#define NSEQS 1280       // NB * NT
#define HID 64           // EMB
#define GATES 256        // 4*HID
#define GSTR 264         // padded sG row stride (floats) to spread LDS banks

typedef __attribute__((ext_vector_type(16))) _Float16 v16h;
typedef __attribute__((ext_vector_type(8)))  _Float16 v8h;
typedef __attribute__((ext_vector_type(8)))  float    v8f;

__device__ __forceinline__ float sigm(float x) { return 1.0f / (1.0f + __expf(-x)); }
// tanh via sigmoid: branch-free, stays on v_exp_f32; saturates to +-1.
__device__ __forceinline__ float ftanh(float x) {
  return __builtin_fmaf(2.0f, sigm(2.0f * x), -1.0f);
}

__device__ __forceinline__ v16h merge16(v8h lo, v8h hi) {
  return __builtin_shufflevector(lo, hi, 0, 1, 2, 3, 4, 5, 6, 7,
                                 8, 9, 10, 11, 12, 13, 14, 15);
}

// A tile 16x32 f16 from row-major memory (row stride `rs` halves, 16B aligned).
// ISA 7.12.2: lanes 0-15 row=lane, halves->K 0..7 / 16..23; lanes 16-31
// row=lane-16, halves->K 8..15 / 24..31.  Two contiguous 16B loads per lane.
__device__ __forceinline__ v16h load_A16(const _Float16* __restrict__ base, int rs) {
  const int lane = threadIdx.x & 31;
  const _Float16* p = base + (lane & 15) * rs + ((lane & 16) ? 8 : 0);
  v8h lo = *(const v8h*)p;
  v8h hi = *(const v8h*)(p + 16);
  return merge16(lo, hi);
}

// Pre-swizzled tile (A or B operand): [lane][16 halves] contiguous.
__device__ __forceinline__ v16h load_pk(const _Float16* __restrict__ base) {
  const int lane = threadIdx.x & 31;
  const _Float16* p = base + lane * 16;
  v8h lo = *(const v8h*)p;
  v8h hi = *(const v8h*)(p + 8);
  return merge16(lo, hi);
}

__device__ __forceinline__ v8f wmma16(v16h a, v16h b, v8f c) {
  return __builtin_amdgcn_wmma_f32_16x16x32_f16(false, a, false, b, (short)0, c, false, false);
}

// B-operand K index for (lane, half):  kloc = 8*(lane>>4) + (half&7) + 16*(half>>3)
// A-operand (lane, half) for (row r, k): lane = r + 16*((k>>3)&1),
//                                        half = (k&7) + 8*((k>>4)&1)

// ---------------------------------------------------------------------------
// lstm1: hidden=4, gates=16.  One thread per (sequence, direction).
// ---------------------------------------------------------------------------
__global__ void k_lstm1(const float* __restrict__ xin, int in_w, int in_stride,
                        const float* __restrict__ Wih_f, const float* __restrict__ Whh_f,
                        const float* __restrict__ bih_f, const float* __restrict__ bhh_f,
                        const float* __restrict__ Wih_b, const float* __restrict__ Whh_b,
                        const float* __restrict__ bih_b, const float* __restrict__ bhh_b,
                        float* __restrict__ out) {
  int tid = blockIdx.x * blockDim.x + threadIdx.x;
  if (tid >= 2 * NSEQS) return;
  int seq = tid >> 1, dir = tid & 1;
  const float* Wih = dir ? Wih_b : Wih_f;
  const float* Whh = dir ? Whh_b : Whh_f;
  const float* bih = dir ? bih_b : bih_f;
  const float* bhh = dir ? bhh_b : bhh_f;
  float h[4] = {0.f, 0.f, 0.f, 0.f};
  float c[4] = {0.f, 0.f, 0.f, 0.f};
  for (int s = 0; s < NT; ++s) {
    int t = dir ? (NT - 1 - s) : s;
    const float* xp = xin + (size_t)(seq * NT + t) * in_stride;
    float g[16];
#pragma unroll 4
    for (int j = 0; j < 16; ++j) {
      float a = bih[j] + bhh[j];
      for (int k = 0; k < in_w; ++k) a += Wih[j * in_w + k] * xp[k];
#pragma unroll
      for (int k = 0; k < 4; ++k) a += Whh[j * 4 + k] * h[k];
      g[j] = a;
    }
#pragma unroll
    for (int k = 0; k < 4; ++k) {
      float cc = sigm(g[4 + k]) * c[k] + sigm(g[k]) * ftanh(g[8 + k]);
      c[k] = cc;
      h[k] = sigm(g[12 + k]) * ftanh(cc);
      out[(size_t)(seq * NT + t) * 8 + dir * 4 + k] = h[k];
    }
  }
}

// pre[(si*NT + j)*4 + p] = mean over dirs of lstm1-layer1 output.  si = b*NT+i.
__global__ void k_pre(const float* __restrict__ h1b, float* __restrict__ preb) {
  int tid = blockIdx.x * blockDim.x + threadIdx.x;
  if (tid >= NSEQS * NT * 4) return;
  int p = tid & 3, ji = tid >> 2;
  const float* src = h1b + (size_t)ji * 8;
  preb[tid] = 0.5f * (src[p] + src[4 + p]);
}

__global__ void k_total(const float* __restrict__ preb, float* __restrict__ totb) {
  int tid = blockIdx.x * blockDim.x + threadIdx.x;
  if (tid >= NSEQS * 4) return;
  int p = tid & 3, si = tid >> 2;
  float s = 0.f;
  for (int j = 0; j < NT; ++j) s += preb[(size_t)(si * NT + j) * 4 + p];
  totb[tid] = s;
}

// feats (f16, padded K=32): featsH[((b*NT+j)*NT + i)*32 + ch]
__global__ void k_featsH(const float* __restrict__ preb, const float* __restrict__ totb,
                         _Float16* __restrict__ featsH) {
  int tid = blockIdx.x * blockDim.x + threadIdx.x;
  if (tid >= NSEQS * NT * 32) return;
  int ch = tid & 31;
  int rowt = tid >> 5;
  int i = rowt % NT;
  int bj = rowt / NT;
  int b = bj / NT, j = bj % NT;
  float v = 0.f;
  if (ch < 8) {
    int p = ch & 3;
    float pv = preb[((size_t)(b * NT + i) * NT + j) * 4 + p];
    v = (ch < 4) ? pv : (totb[(b * NT + i) * 4 + p] - pv);
  }
  featsH[tid] = (_Float16)v;
}

// ---------------------------------------------------------------------------
// Repack one lstm2 (layer,dir) unit into WMMA B-operand register layout:
// P[((kk*16 + nt)*32 + lane)*16 + half] = WT[kloc][n],
//   kloc = kk*32 + 8*(lane>>4) + (half&7) + 16*(half>>3),  n = nt*16 + (lane&15),
//   WT[k][n] = W[n][k]  (W row-major (256, in_w) / (256, 64)).
// ---------------------------------------------------------------------------
__global__ void k_prepw(const float* __restrict__ Wih, const float* __restrict__ Whh,
                        const float* __restrict__ bih, const float* __restrict__ bhh,
                        int in_w, int KIN,
                        _Float16* __restrict__ WihP, _Float16* __restrict__ WhhP,
                        float* __restrict__ biasC) {
  int tid = blockIdx.x * blockDim.x + threadIdx.x;
  if (tid < KIN * GATES) {
    int half = tid & 15, lane = (tid >> 4) & 31, nt = (tid >> 9) & 15, kk = tid >> 13;
    int k = kk * 32 + 8 * (lane >> 4) + (half & 7) + 16 * (half >> 3);
    int n = nt * 16 + (lane & 15);
    WihP[tid] = (k < in_w) ? (_Float16)Wih[n * in_w + k] : (_Float16)0.0f;
  } else {
    int t2 = tid - KIN * GATES;
    if (t2 < HID * GATES) {
      int half = t2 & 15, lane = (t2 >> 4) & 31, nt = (t2 >> 9) & 15, kk = t2 >> 13;
      int k = kk * 32 + 8 * (lane >> 4) + (half & 7) + 16 * (half >> 3);
      int n = nt * 16 + (lane & 15);
      WhhP[t2] = (_Float16)Whh[n * HID + k];
    } else if (t2 - HID * GATES < GATES) {
      int n = t2 - HID * GATES;
      biasC[n] = bih[n] + bhh[n];
    }
  }
}

// ---------------------------------------------------------------------------
// lstm2 recurrence, fused input projection.  16 sequences / block, 4 waves,
// grid = (80, 2) with blockIdx.y = direction.  Whh B-tiles (and Wih for the
// K=32 layer) live in VGPRs for the whole time loop; per step only the h tile
// (2x ds_load_b128) and activations are fetched.
// ---------------------------------------------------------------------------
template <int KIN>
__global__ __launch_bounds__(128) void k_lstm2_rec(
    const _Float16* __restrict__ xin,                       // [seq][NT][KIN] f16
    const _Float16* __restrict__ WihP_f, const _Float16* __restrict__ WihP_b,
    const _Float16* __restrict__ WhhP_f, const _Float16* __restrict__ WhhP_b,
    const float* __restrict__ bias_f,   const float* __restrict__ bias_b,
    _Float16* __restrict__ oout,                            // [seq][NT][128] f16
    float* __restrict__ hn, float* __restrict__ cn)         // [2][NSEQS][HID]
{
  __shared__ _Float16 sHp[2 * 32 * 16];    // h in A-operand layout, 2 KB
  __shared__ float    sC[16 * HID];        // 4 KB
  __shared__ float    sG[16 * GSTR];       // ~16.5 KB
  __shared__ float    sBias[GATES];

  const int tid  = threadIdx.x;
  const int wave = tid >> 5;
  const int lane = tid & 31;
  const int seq0 = blockIdx.x * 16;
  const int dir  = blockIdx.y;

  const _Float16* WihP = dir ? WihP_b : WihP_f;
  const _Float16* WhhP = dir ? WhhP_b : WhhP_f;
  const float*    bias = dir ? bias_b : bias_f;
  float* hnd = hn + (size_t)dir * NSEQS * HID;
  float* cnd = cn + (size_t)dir * NSEQS * HID;
  const int dirofs = dir * HID;

  for (int i = tid; i < GATES; i += 128) sBias[i] = bias[i];
  for (int i = tid; i < 2 * 32 * 16; i += 128) sHp[i] = (_Float16)0.0f;
  for (int i = tid; i < 16 * HID; i += 128) sC[i] = 0.0f;
  __syncthreads();

  // hoist time-invariant weight B-tiles into VGPRs
  v16h bwh0[4], bwh1[4];
#pragma unroll
  for (int nt = 0; nt < 4; ++nt) {
    bwh0[nt] = load_pk(WhhP + (wave * 4 + nt) * 512);
    bwh1[nt] = load_pk(WhhP + (16 + wave * 4 + nt) * 512);
  }
  v16h bwi0[4];
  if constexpr (KIN == 32) {
#pragma unroll
    for (int nt = 0; nt < 4; ++nt)
      bwi0[nt] = load_pk(WihP + (wave * 4 + nt) * 512);
  }

  const int col = lane & 15;
  const int rb  = (lane & 16) ? 8 : 0;

  for (int s = 0; s < NT; ++s) {
    const int t = dir ? (NT - 1 - s) : s;
    const _Float16* xbase = xin + ((size_t)seq0 * NT + t) * KIN;

    // prefetch next timestep's activation rows (global_prefetch_b8)
    if (s + 1 < NT) {
      const int t2 = dir ? (NT - 2 - s) : (s + 1);
      __builtin_prefetch(xin + ((size_t)(seq0 + (lane & 15)) * NT + t2) * KIN, 0, 0);
    }

    v16h ain[KIN / 32];
#pragma unroll
    for (int kk = 0; kk < KIN / 32; ++kk)
      ain[kk] = load_A16(xbase + kk * 32, NT * KIN);
    v16h ah0 = load_pk(sHp);
    v16h ah1 = load_pk(sHp + 512);

#pragma unroll
    for (int nt = 0; nt < 4; ++nt) {
      const int n0 = wave * 64 + nt * 16;
      const int ntile = wave * 4 + nt;
      v8f acc;
      const float bv = sBias[n0 + col];
#pragma unroll
      for (int i = 0; i < 8; ++i) acc[i] = bv;
      if constexpr (KIN == 32) {
        acc = wmma16(ain[0], bwi0[nt], acc);
      } else {
#pragma unroll
        for (int kk = 0; kk < KIN / 32; ++kk)
          acc = wmma16(ain[kk], load_pk(WihP + (kk * 16 + ntile) * 512), acc);
      }
      acc = wmma16(ah0, bwh0[nt], acc);
      acc = wmma16(ah1, bwh1[nt], acc);
#pragma unroll
      for (int i = 0; i < 8; ++i) sG[(rb + i) * GSTR + n0 + col] = acc[i];
    }
    __syncthreads();

    // gate nonlinearity + state update: 16x64 elems, 8 per thread.
    // h written straight back in A-operand layout.
#pragma unroll
    for (int e = 0; e < 8; ++e) {
      const int idx = tid + e * 128;
      const int r = idx >> 6, k = idx & 63;
      const float* gr = sG + r * GSTR;
      float cc = sigm(gr[64 + k]) * sC[r * HID + k] + sigm(gr[k]) * ftanh(gr[128 + k]);
      float hh = sigm(gr[192 + k]) * ftanh(cc);
      sC[r * HID + k] = cc;
      const int alane = r + 16 * ((k >> 3) & 1);
      const int ahalf = (k & 7) + 8 * ((k >> 4) & 1);
      sHp[(k >> 5) * 512 + alane * 16 + ahalf] = (_Float16)hh;
      oout[((size_t)(seq0 + r) * NT + t) * 128 + dirofs + k] = (_Float16)hh;
      if (s == NT - 1) {
        hnd[(seq0 + r) * HID + k] = hh;
        cnd[(seq0 + r) * HID + k] = cc;
      }
    }
    __syncthreads();
  }
}

// o_red[b,j,e] = mean over (i, dir) of o1[(b*NT+j), i, dir*64+e]
__global__ void k_ored(const _Float16* __restrict__ o1h, float* __restrict__ ored) {
  int tid = blockIdx.x * blockDim.x + threadIdx.x;
  if (tid >= NSEQS * HID) return;
  int row = tid >> 6, e = tid & 63;
  const _Float16* p = o1h + (size_t)row * NT * 128 + e;
  float s = 0.f;
  for (int i = 0; i < NT; ++i) s += (float)p[i * 128] + (float)p[i * 128 + 64];
  ored[tid] = s * (1.0f / (2.0f * NT));
}

// z[(b2*NT+j)*576 + c]: reproduces the reference flat_states() reshape shuffle.
__global__ void k_z(const float* __restrict__ hn, const float* __restrict__ cn,
                    const float* __restrict__ ored, float* __restrict__ z) {
  int tid = blockIdx.x * blockDim.x + threadIdx.x;
  if (tid >= NSEQS * 576) return;
  int r = tid / 576, c = tid % 576;
  int b2 = r / NT, j = r % NT;
  float v;
  if (c < 512) {
    const float* src = (c < 256) ? hn : cn;
    int k = c & 255;
    int l = b2 >> 1;
    int b = ((b2 & 1) << 2) | (k >> 6);
    int e = k & 63;
    v = src[((size_t)l * NSEQS + (b * NT + j)) * HID + e];
  } else {
    v = ored[(size_t)(b2 * NT + j) * HID + (c - 512)];
  }
  z[tid] = v;
}

// Fused 5-layer MLP + sigmoid head + reparameterization.  One block per row.
__global__ __launch_bounds__(256) void k_mlp(
    const float* __restrict__ z,
    const float* __restrict__ W0, const float* __restrict__ b0,
    const float* __restrict__ W1, const float* __restrict__ b1,
    const float* __restrict__ W2, const float* __restrict__ b2,
    const float* __restrict__ W3, const float* __restrict__ b3,
    const float* __restrict__ W4, const float* __restrict__ b4,
    const float* __restrict__ eps, float* __restrict__ out) {
  __shared__ float buf0[576];
  __shared__ float buf1[256];
  const int r = blockIdx.x, tid = threadIdx.x;
  for (int i = tid; i < 576; i += 256) buf0[i] = z[(size_t)r * 576 + i];
  __syncthreads();
  if (tid < 256) {                                   // 576 -> 256, relu
    float a = b0[tid]; const float* w = W0 + (size_t)tid * 576;
    for (int k = 0; k < 576; ++k) a += w[k] * buf0[k];
    buf1[tid] = fmaxf(a, 0.f);
  }
  __syncthreads();
  if (tid < 128) {                                   // 256 -> 128, relu
    float a = b1[tid]; const float* w = W1 + (size_t)tid * 256;
    for (int k = 0; k < 256; ++k) a += w[k] * buf1[k];
    buf0[tid] = fmaxf(a, 0.f);
  }
  __syncthreads();
  if (tid < 64) {                                    // 128 -> 64, relu
    float a = b2[tid]; const float* w = W2 + (size_t)tid * 128;
    for (int k = 0; k < 128; ++k) a += w[k] * buf0[k];
    buf1[tid] = fmaxf(a, 0.f);
  }
  __syncthreads();
  if (tid < 32) {                                    // 64 -> 32, relu
    float a = b3[tid]; const float* w = W3 + (size_t)tid * 64;
    for (int k = 0; k < 64; ++k) a += w[k] * buf1[k];
    buf0[tid] = fmaxf(a, 0.f);
  }
  __syncthreads();
  if (tid < 64) {                                    // 32 -> 64, sigmoid
    float a = b4[tid]; const float* w = W4 + (size_t)tid * 32;
    for (int k = 0; k < 32; ++k) a += w[k] * buf0[k];
    buf1[tid] = sigm(a);
  }
  __syncthreads();
  if (tid < 32) {
    const float mu = buf1[tid], lv = buf1[32 + tid];
    const float sd = __expf(0.5f * lv);
    const float ev = eps[(size_t)r * 32 + tid];
    const size_t o = (size_t)r * 32 + tid;
    out[o] = mu + ev * sd;
    out[(size_t)NSEQS * 32 + o] = mu;
    out[(size_t)2 * NSEQS * 32 + o] = lv;
  }
}

// ---------------------------------------------------------------------------
extern "C" void kernel_launch(void* const* d_in, const int* in_sizes, int n_in,
                              void* d_out, int out_size, void* d_ws, size_t ws_size,
                              hipStream_t stream) {
  (void)in_sizes; (void)n_in; (void)out_size; (void)ws_size;
  const float* x = (const float*)d_in[0];
  const float* eps = (const float*)d_in[43];
  float* out = (float*)d_out;

  // --- workspace carve-out ---
  char* w = (char*)d_ws;
  size_t off = 0;
  auto take = [&](size_t bytes) -> void* {
    void* p = w + off;
    off = (off + bytes + 255) & ~(size_t)255;
    return p;
  };
  float*    h1a    = (float*)take(sizeof(float) * (size_t)NSEQS * NT * 8);
  float*    h1b    = (float*)take(sizeof(float) * (size_t)NSEQS * NT * 8);
  float*    preb   = (float*)take(sizeof(float) * (size_t)NSEQS * NT * 4);
  float*    totb   = (float*)take(sizeof(float) * (size_t)NSEQS * 4);
  _Float16* featsH = (_Float16*)take(sizeof(_Float16) * (size_t)NSEQS * NT * 32);
  _Float16* WihP0  = (_Float16*)take(sizeof(_Float16) * 2 * 32 * GATES);
  _Float16* WihP1  = (_Float16*)take(sizeof(_Float16) * 2 * 128 * GATES);
  _Float16* WhhP   = (_Float16*)take(sizeof(_Float16) * 4 * HID * GATES);
  float*    biasC  = (float*)take(sizeof(float) * 4 * GATES);
  _Float16* o0h    = (_Float16*)take(sizeof(_Float16) * (size_t)NSEQS * NT * 128);
  _Float16* o1h    = (_Float16*)take(sizeof(_Float16) * (size_t)NSEQS * NT * 128);
  float*    hn     = (float*)take(sizeof(float) * 4 * (size_t)NSEQS * HID);
  float*    cn     = (float*)take(sizeof(float) * 4 * (size_t)NSEQS * HID);
  float*    ored   = (float*)take(sizeof(float) * (size_t)NSEQS * HID);
  float*    zb     = (float*)take(sizeof(float) * (size_t)NSEQS * 576);

  auto F = [&](int i) { return (const float*)d_in[i]; };

  // --- stage 1: lstm1 (H=4, VALU) ---
  k_lstm1<<<(2 * NSEQS + 127) / 128, 128, 0, stream>>>(
      x, 1, 1, F(1), F(2), F(3), F(4), F(5), F(6), F(7), F(8), h1a);
  k_lstm1<<<(2 * NSEQS + 127) / 128, 128, 0, stream>>>(
      h1a, 8, 8, F(9), F(10), F(11), F(12), F(13), F(14), F(15), F(16), h1b);

  k_pre<<<(NSEQS * NT * 4 + 255) / 256, 256, 0, stream>>>(h1b, preb);
  k_total<<<(NSEQS * 4 + 255) / 256, 256, 0, stream>>>(preb, totb);
  k_featsH<<<(NSEQS * NT * 32 + 255) / 256, 256, 0, stream>>>(preb, totb, featsH);

  // --- repack lstm2 weights into packed WMMA B-operand layout (f16) ---
  for (int u = 0; u < 4; ++u) {
    const int layer = u >> 1;
    const int in_w = layer ? 128 : 8;
    const int KIN = layer ? 128 : 32;
    _Float16* wt = layer ? (WihP1 + (size_t)(u - 2) * 128 * GATES)
                         : (WihP0 + (size_t)u * 32 * GATES);
    const int base = 17 + 4 * u;
    const int tot = KIN * GATES + HID * GATES + GATES;
    k_prepw<<<(tot + 255) / 256, 256, 0, stream>>>(
        F(base), F(base + 1), F(base + 2), F(base + 3),
        in_w, KIN, wt, WhhP + (size_t)u * HID * GATES, biasC + u * GATES);
  }

  // --- stage 2: lstm2 WMMA recurrences (both directions in grid.y) ---
  dim3 grec(NSEQS / 16, 2);
  k_lstm2_rec<32><<<grec, 128, 0, stream>>>(
      featsH, WihP0, WihP0 + 32 * GATES, WhhP, WhhP + (size_t)HID * GATES,
      biasC, biasC + GATES, o0h, hn, cn);
  k_lstm2_rec<128><<<grec, 128, 0, stream>>>(
      o0h, WihP1, WihP1 + 128 * GATES,
      WhhP + (size_t)2 * HID * GATES, WhhP + (size_t)3 * HID * GATES,
      biasC + 2 * GATES, biasC + 3 * GATES, o1h,
      hn + (size_t)2 * NSEQS * HID, cn + (size_t)2 * NSEQS * HID);

  // --- head ---
  k_ored<<<(NSEQS * HID + 255) / 256, 256, 0, stream>>>(o1h, ored);
  k_z<<<(NSEQS * 576 + 255) / 256, 256, 0, stream>>>(hn, cn, ored, zb);
  k_mlp<<<NSEQS, 256, 0, stream>>>(
      zb, F(33), F(34), F(35), F(36), F(37), F(38), F(39), F(40), F(41), F(42),
      eps, out);
}
